// GraphKMeans_70875550319049
// MI455X (gfx1250) — matrix operand order repaired and checked by reading, not compiled, not run on hardware
//
#include <hip/hip_runtime.h>
#include <math.h>

// ---------------- types for gfx1250 WMMA ----------------
typedef __attribute__((ext_vector_type(16))) __bf16 v16bf;
typedef __attribute__((ext_vector_type(8)))  __bf16 v8bf;
typedef __attribute__((ext_vector_type(8)))  float  v8f;

#define D_DIM 256
#define K_DIM 64

// ---------------- helpers ----------------
__device__ __forceinline__ unsigned int f_as_u(float x) { return __builtin_bit_cast(unsigned int, x); }
__device__ __forceinline__ float u_as_f(unsigned int u) { return __builtin_bit_cast(float, u); }
__device__ __forceinline__ __bf16 u16_as_bf(unsigned short h) { return __builtin_bit_cast(__bf16, h); }

// Split f32 into bf16 hi (truncated) + bf16 lo (residual). hi+lo carries ~16
// mantissa bits; products accumulate in f32 inside the WMMA, so dropping the
// lo*lo term leaves ~2^-16 relative error on the cross term.
__device__ __forceinline__ void split_bf16(float x, __bf16& hi, __bf16& lo) {
    unsigned int u = f_as_u(x);
    hi = u16_as_bf((unsigned short)(u >> 16));
    float hif = u_as_f(u & 0xffff0000u);
    float r = x - hif;
    lo = u16_as_bf((unsigned short)(f_as_u(r) >> 16));
}

// ---------------- kernel 1: centroid prep (c^2 + bf16 hi/lo split) ----------------
__global__ __launch_bounds__(D_DIM)
void kmeans_prep(const float* __restrict__ cent, float* __restrict__ csq,
                 __bf16* __restrict__ chi, __bf16* __restrict__ clo) {
    __shared__ float sm[D_DIM];
    const int k = blockIdx.x;      // 0..63
    const int d = threadIdx.x;     // 0..255
    float v = cent[k * D_DIM + d];
    __bf16 h, l;
    split_bf16(v, h, l);
    chi[k * D_DIM + d] = h;
    clo[k * D_DIM + d] = l;
    sm[d] = v * v;
    __syncthreads();
    for (int off = 128; off > 0; off >>= 1) {
        if (d < off) sm[d] += sm[d + off];
        __syncthreads();
    }
    if (d == 0) csq[k] = sm[0];
}

// ---------------- kernel 2: main — cross term via split-bf16 WMMA + fused soft-min ----------------
__global__ __launch_bounds__(256)
void kmeans_main(const float* __restrict__ emb,
                 const int* __restrict__ alpha_p,
                 const float* __restrict__ csq,
                 const __bf16* __restrict__ chi,
                 const __bf16* __restrict__ clo,
                 float* __restrict__ partial, int Ncols) {
    __shared__ float s_csq[K_DIM];
    if (threadIdx.x < K_DIM) s_csq[threadIdx.x] = csq[threadIdx.x];
    __syncthreads();

    const int lane   = threadIdx.x & 31;
    const int waveId = blockIdx.x * 8 + (threadIdx.x >> 5);
    const int n0     = waveId * 16;          // 16 embedding columns per wave
    float val = 0.0f;

    if (n0 < Ncols) {                         // wave-uniform: EXEC stays all-ones for WMMA
        const int col = lane & 15;            // matrix column N (embedding index within tile)
        const int hh  = lane >> 4;            // 0: lanes 0-15, 1: lanes 16-31
        const float alphaf = (float)alpha_p[0];

        // B fragment source: lane reads 16 consecutive f32 of its column.
        // lanes 0-15 -> K(inner d) = d0+0..15 ; lanes 16-31 -> d0+16..31  (contiguous k order, same as A)
        const float* embRow = emb + (size_t)(n0 + col) * D_DIM + hh * 16;
        // A fragment half-slot offset: lanes 0-15 slots: d0+0..7 / d0+8..15,
        // lanes 16-31: d0+16..23 / d0+24..31  (contiguous, matches B)
        const int aoff = hh * 16;

        v8f acc0 = {}, acc1 = {}, acc2 = {}, acc3 = {};
        float xsq = 0.0f;

        #pragma unroll
        for (int ch = 0; ch < 8; ++ch) {       // 8 chunks of 32 over D=256
            // ----- B fragment: load 64B/lane, split to bf16 hi/lo, accumulate x^2 -----
            const float4* bp = (const float4*)(embRow + ch * 32);
            float4 b0 = bp[0], b1 = bp[1], b2 = bp[2], b3 = bp[3];
            float bv[16] = { b0.x, b0.y, b0.z, b0.w,  b1.x, b1.y, b1.z, b1.w,
                             b2.x, b2.y, b2.z, b2.w,  b3.x, b3.y, b3.z, b3.w };
            v16bf bhi, blo;
            #pragma unroll
            for (int i = 0; i < 16; ++i) {
                __bf16 h, l;
                split_bf16(bv[i], h, l);
                bhi[i] = h; blo[i] = l;
                xsq = fmaf(bv[i], bv[i], xsq);
            }

            // ----- 4 k-tiles: pre-split A fragments load straight into layout -----
            #pragma unroll
            for (int t = 0; t < 4; ++t) {
                const size_t arow = (size_t)(t * 16 + col) * D_DIM + (size_t)(ch * 32 + aoff);
                v8bf h0 = __builtin_bit_cast(v8bf, *(const uint4*)(chi + arow));
                v8bf h1 = __builtin_bit_cast(v8bf, *(const uint4*)(chi + arow + 8));
                v8bf l0 = __builtin_bit_cast(v8bf, *(const uint4*)(clo + arow));
                v8bf l1 = __builtin_bit_cast(v8bf, *(const uint4*)(clo + arow + 8));
                v16bf ahi, alo;
                #pragma unroll
                for (int i = 0; i < 8; ++i) {
                    ahi[i] = h0[i]; ahi[8 + i] = h1[i];
                    alo[i] = l0[i]; alo[8 + i] = l1[i];
                }
                v8f acc = (t == 0) ? acc0 : (t == 1) ? acc1 : (t == 2) ? acc2 : acc3;
                acc = __builtin_amdgcn_wmma_f32_16x16x32_bf16(false, ahi, false, bhi, (short)0, acc, false, false);
                acc = __builtin_amdgcn_wmma_f32_16x16x32_bf16(false, ahi, false, blo, (short)0, acc, false, false);
                acc = __builtin_amdgcn_wmma_f32_16x16x32_bf16(false, alo, false, bhi, (short)0, acc, false, false);
                if      (t == 0) acc0 = acc;
                else if (t == 1) acc1 = acc;
                else if (t == 2) acc2 = acc;
                else             acc3 = acc;
            }
        }

        // ----- epilogue: s_k = c^2_k - 2*cross ; soft-min over all 64 k -----
        // C layout: VGPR j <-> row M = 8*hh + j per k-tile; column N = col.
        const int kb = hh * 8;
        float sv[32];
        #pragma unroll
        for (int j = 0; j < 8; ++j) {
            sv[j]      = s_csq[ 0 + kb + j] - 2.0f * acc0[j];
            sv[8 + j]  = s_csq[16 + kb + j] - 2.0f * acc1[j];
            sv[16 + j] = s_csq[32 + kb + j] - 2.0f * acc2[j];
            sv[24 + j] = s_csq[48 + kb + j] - 2.0f * acc3[j];
        }
        float mn = 3.402823466e+38f;
        #pragma unroll
        for (int i = 0; i < 32; ++i) mn = fminf(mn, sv[i]);
        mn = fminf(mn, __shfl_xor(mn, 16, 32));          // merge lane pair (full 64 k)

        float se = 0.0f, num = 0.0f;
        #pragma unroll
        for (int i = 0; i < 32; ++i) {
            float e = __expf(-alphaf * (sv[i] - mn));     // stable softmax over -alpha*s
            se += e;
            num = fmaf(e, sv[i], num);
        }
        se  += __shfl_xor(se, 16, 32);
        num += __shfl_xor(num, 16, 32);
        xsq += __shfl_xor(xsq, 16, 32);                   // full sum_d x^2 for this column

        float contrib = num / se + xsq;                   // sum_k p_k d_k  (x^2 re-added)
        val = (lane < 16) ? contrib : 0.0f;               // avoid double count
        #pragma unroll
        for (int off = 16; off > 0; off >>= 1) val += __shfl_xor(val, off, 32);
    }
    if (lane == 0) partial[waveId] = val;                 // unique slot per wave: deterministic
}

// ---------------- kernel 3: deterministic final reduction ----------------
__global__ __launch_bounds__(256)
void kmeans_reduce(const float* __restrict__ partial, int numWaves,
                   float* __restrict__ out, float scale) {
    __shared__ float sm[256];
    float s = 0.0f;
    for (int i = threadIdx.x; i < numWaves; i += 256) s += partial[i];
    sm[threadIdx.x] = s;
    __syncthreads();
    for (int off = 128; off > 0; off >>= 1) {
        if (threadIdx.x < off) sm[threadIdx.x] += sm[threadIdx.x + off];
        __syncthreads();
    }
    if (threadIdx.x == 0) out[0] = sm[0] * scale;   // LAMBD * mean
}

// ---------------- host launcher ----------------
extern "C" void kernel_launch(void* const* d_in, const int* in_sizes, int n_in,
                              void* d_out, int out_size, void* d_ws, size_t ws_size,
                              hipStream_t stream) {
    const float* emb  = (const float*)d_in[0];
    const float* cent = (const float*)d_in[1];
    const int*   alp  = (const int*)d_in[2];
    float* out = (float*)d_out;

    const int N = in_sizes[0] / D_DIM;   // 200000

    // workspace layout (bytes): [0,256) csq ; [256, +32K) cent hi bf16 ;
    // [+32K, +64K) cent lo bf16 ; then per-wave partials (f32)
    char* ws = (char*)d_ws;
    float*  csq     = (float*)ws;
    __bf16* chi     = (__bf16*)(ws + 256);
    __bf16* clo     = (__bf16*)(ws + 256 + (size_t)K_DIM * D_DIM * 2);
    float*  partial = (float*)(ws + 256 + 2 * (size_t)K_DIM * D_DIM * 2);

    const int waves    = (N + 15) / 16;       // 16 columns per wave
    const int blocks   = (waves + 7) / 8;     // 8 waves per 256-thread block
    const int numWaves = blocks * 8;          // inactive waves write 0

    kmeans_prep  <<<K_DIM, D_DIM, 0, stream>>>(cent, csq, chi, clo);
    kmeans_main  <<<blocks, 256, 0, stream>>>(emb, alp, csq, chi, clo, partial, N);
    kmeans_reduce<<<1, 256, 0, stream>>>(partial, numWaves, out, 1.0f / (float)N);
}
